// PointCloudEncoder_53017076302431
// MI455X (gfx1250) — matrix operand order repaired and checked by reading, not compiled, not run on hardware
//
#include <hip/hip_runtime.h>

typedef __attribute__((ext_vector_type(16))) _Float16 v16h;
typedef __attribute__((ext_vector_type(8)))  _Float16 v8h;
typedef __attribute__((ext_vector_type(8)))  float    v8f;

#define NPTS 20480
#define NB   8
#define NK1  512
#define NS1  256
#define NK2  64
#define NS2  32

// f16 weight-buffer offsets (in halfs), K padded to multiples of 32
constexpr int WO1 = 0;       // 64 x 32   (cin 8  -> pad 32)
constexpr int WO2 = 2048;    // 64 x 64
constexpr int WO3 = 6144;    // 256 x 64
constexpr int WO4 = 22528;   // 256 x 288 (cin 259 -> pad 288)
constexpr int WO5 = 96256;   // 256 x 256
constexpr int WO6 = 161792;  // 512 x 256
constexpr int WO7 = 292864;  // 512 x 512
constexpr int WO8 = 555008;  // 512 x 512
// scale/shift offsets (floats)
constexpr int SO1 = 0, SO2 = 64, SO3 = 128, SO4 = 384, SO5 = 640, SO6 = 896, SO7 = 1408, SO8 = 1920;

// d_out offsets (floats): xyz1, xyz2, f1, f2, gf concatenated
constexpr size_t OUT_XYZ1 = 0;
constexpr size_t OUT_XYZ2 = 12288;
constexpr size_t OUT_F1   = 13824;
constexpr size_t OUT_F2   = 1062400;
constexpr size_t OUT_GF   = 1324544;

// workspace offsets (bytes)
constexpr size_t WS_W    = 0;        // 817152 halfs = 1634304 B
constexpr size_t WS_SC   = 1634304;  // 2432 floats
constexpr size_t WS_SH   = 1644032;  // 2432 floats
constexpr size_t WS_IDX1 = 1653760;  // 8*512*256 ints = 4 MB
constexpr size_t WS_IDX2 = 5848064;  // 8*64*32 ints
constexpr size_t WS_F1H  = 5913600;  // 8*256*512 halfs = 2 MB
constexpr size_t WS_F2H  = 8010752;  // 8*64*512 halfs

// ---------------- WMMA helpers (CDNA5 16x16x32 f16, f32 accum) ----------------

__device__ __forceinline__ v8f wmma_f16(v16h a, v16h b, v8f c) {
  return __builtin_amdgcn_wmma_f32_16x16x32_f16(false, a, false, b, (short)0, c, false, false);
}

// A-matrix 16x32 f16 tile: ISA layout — lane m=lane&15, h=lane>>4,
// halfs j<8 -> K = kbase + h*8 + j ; j>=8 -> K = kbase + 16 + h*8 + (j-8)
__device__ __forceinline__ v16h load_a16(const _Float16* base, int row0, int strideH, int kbase, int lane) {
  const int m = lane & 15, h = lane >> 4;
  const _Float16* p = base + (size_t)(row0 + m) * strideH + kbase + h * 8;
  v8h lo = *(const v8h*)(p);
  v8h hi = *(const v8h*)(p + 16);
  v16h r;
#pragma unroll
  for (int i = 0; i < 8; ++i) { r[i] = lo[i]; r[i + 8] = hi[i]; }
  return r;
}

// B-matrix 32x16 f16 tile: lane n=lane&15, h=lane>>4, halfs j<8 -> K = kbase + h*8 + j ; j>=8 -> K = kbase + 16 + h*8 + (j-8)
// Source stored [point][channel] (row = N/point, contiguous K) -> 2x b128 per lane.
__device__ __forceinline__ v16h load_b16(const _Float16* base, int col0, int strideH, int kbase, int lane) {
  const int n = lane & 15, h = lane >> 4;
  const _Float16* p = base + (size_t)(col0 + n) * strideH + kbase + h * 8;
  v8h lo = *(const v8h*)(p);
  v8h hi = *(const v8h*)(p + 16);
  v16h r;
#pragma unroll
  for (int i = 0; i < 8; ++i) { r[i] = lo[i]; r[i + 8] = hi[i]; }
  return r;
}

// C/D 16x16 f32: VGPR r, lanes0-15 -> M=r, lanes16-31 -> M=r+8; N=lane&15
__device__ __forceinline__ void store_bn_relu_f16(_Float16* dst, int strideH, int nbase, int mbase,
                                                  v8f acc, const float* sc, const float* sh, int lane) {
  const int n = lane & 15, h = lane >> 4;
#pragma unroll
  for (int r = 0; r < 8; ++r) {
    int m = mbase + r + 8 * h;
    float y = fmaxf(acc[r] * sc[m] + sh[m], 0.f);
    dst[(size_t)(nbase + n) * strideH + m] = (_Float16)y;
  }
}

__device__ __forceinline__ float angle3(float ax, float ay, float az, float bx, float by, float bz) {
  float cx = ay * bz - az * by;
  float cy = az * bx - ax * bz;
  float cz = ax * by - ay * bx;
  float s2 = cx * cx + cy * cy + cz * cz;
  float cn = s2 > 0.f ? sqrtf(s2) : 0.f;
  float dp = ax * bx + ay * by + az * bz;
  return atan2f(cn, dp);
}

// ---------------- param prep: fold conv bias + BN, convert W to f16 with K pad ----------------

__global__ __launch_bounds__(256) void prep_layer(const float* __restrict__ W, const float* __restrict__ bias,
                                                  const float* __restrict__ beta, const float* __restrict__ gamma,
                                                  const float* __restrict__ mean, const float* __restrict__ var,
                                                  _Float16* __restrict__ wout, float* __restrict__ sc,
                                                  float* __restrict__ sh, int cout, int cin, int cinPad) {
  int i = blockIdx.x * 256 + threadIdx.x;
  int tot = cout * cinPad;
  if (i < tot) {
    int o = i / cinPad, c = i % cinPad;
    float v = (c < cin) ? W[(size_t)o * cin + c] : 0.f;
    wout[i] = (_Float16)v;
  }
  if (i < cout) {
    float s = gamma[i] * rsqrtf(var[i] + 1e-5f);
    sc[i] = s;
    sh[i] = (bias[i] - mean[i]) * s + beta[i];
  }
}

// ---------------- farthest point sampling: 1 workgroup / batch, points in registers ----------------

template <int NT, int PPT>
__global__ void fps_kernel(const float* __restrict__ pc, int n, int k, float* __restrict__ out) {
  const int b = blockIdx.x, t = threadIdx.x;
  const float* px = pc + (size_t)b * 3 * n;
  const float* py = px + n;
  const float* pz = py + n;
  float lx[PPT], ly[PPT], lz[PPT], ld[PPT];
#pragma unroll
  for (int j = 0; j < PPT; ++j) {
    int i = t + j * NT;
    bool ok = i < n;
    lx[j] = ok ? px[i] : 0.f;
    ly[j] = ok ? py[i] : 0.f;
    lz[j] = ok ? pz[i] : 0.f;
    ld[j] = ok ? 1e10f : -1e30f;
  }
  __shared__ float cen[3];
  __shared__ float rv[NT];
  __shared__ int ri[NT];
  int far = 0;
  for (int it = 0; it < k; ++it) {
    if (t == 0) { cen[0] = px[far]; cen[1] = py[far]; cen[2] = pz[far]; }
    __syncthreads();
    float cx = cen[0], cy = cen[1], cz = cen[2];
    if (t == 0) {
      out[((size_t)b * 3 + 0) * k + it] = cx;
      out[((size_t)b * 3 + 1) * k + it] = cy;
      out[((size_t)b * 3 + 2) * k + it] = cz;
    }
    float bv = -1e30f;
    int bi = 0;
#pragma unroll
    for (int j = 0; j < PPT; ++j) {
      float dx = lx[j] - cx, dy = ly[j] - cy, dz = lz[j] - cz;
      float d = dx * dx + dy * dy + dz * dz;
      ld[j] = fminf(ld[j], d);
      if (ld[j] > bv) { bv = ld[j]; bi = t + j * NT; }
    }
    rv[t] = bv;
    ri[t] = bi;
    __syncthreads();
    for (int s = NT / 2; s > 0; s >>= 1) {
      if (t < s && rv[t + s] > rv[t]) { rv[t] = rv[t + s]; ri[t] = ri[t + s]; }
      __syncthreads();
    }
    far = ri[0];
    __syncthreads();
  }
}

// ---------------- kNN stage 1: 256-NN of 20480, register-resident distances + bit binary search ----------------

__global__ __launch_bounds__(256) void knn1_kernel(const float* __restrict__ xyz, const float* __restrict__ kpts,
                                                   int* __restrict__ idxout) {
  const int k = blockIdx.x, b = blockIdx.y, t = threadIdx.x;
  __shared__ float rv[256];
  __shared__ int ri[256];
  __shared__ int cA, cB;
  const float* px = xyz + (size_t)b * 3 * NPTS;
  const float* py = px + NPTS;
  const float* pz = py + NPTS;
  float kx = kpts[(size_t)b * 3 * NK1 + k];
  float ky = kpts[(size_t)b * 3 * NK1 + NK1 + k];
  float kz = kpts[(size_t)b * 3 * NK1 + 2 * NK1 + k];
  float ld[80];
  float bv = 3.0e38f;
  int bi = 0;
#pragma unroll
  for (int j = 0; j < 80; ++j) {
    int i = t + j * 256;
    float dx = px[i] - kx, dy = py[i] - ky, dz = pz[i] - kz;
    float d = dx * dx + dy * dy + dz * dz;
    ld[j] = d;
    if (d < bv) { bv = d; bi = i; }
  }
  rv[t] = bv;
  ri[t] = bi;
  __syncthreads();
  for (int s = 128; s > 0; s >>= 1) {
    if (t < s && rv[t + s] < rv[t]) { rv[t] = rv[t + s]; ri[t] = ri[t + s]; }
    __syncthreads();
  }
  const int n0 = ri[0];
  const int base = ((b * NK1) + k) * NS1;
  if (t == 0) idxout[base] = n0;
  {  // remove chosen nearest (avoid dynamic register index)
    int j0 = n0 >> 8;
    bool mine = (n0 & 255) == t;
#pragma unroll
    for (int j = 0; j < 80; ++j)
      if (mine && j == j0) ld[j] = 3.0e38f;
  }
  __syncthreads();
  // smallest T with count(bits <= T) >= 255 (uint order == float order for d >= 0)
  unsigned lo = 0u, hi = 0x7F800000u;
  while (lo < hi) {
    unsigned mid = lo + ((hi - lo) >> 1);
    int c = 0;
#pragma unroll
    for (int j = 0; j < 80; ++j) c += (__float_as_uint(ld[j]) <= mid) ? 1 : 0;
    ri[t] = c;
    __syncthreads();
    for (int s = 128; s > 0; s >>= 1) {
      if (t < s) ri[t] += ri[t + s];
      __syncthreads();
    }
    int tot = ri[0];
    __syncthreads();
    if (tot >= 255) hi = mid; else lo = mid + 1;
  }
  const unsigned T = lo;
  if (t == 0) { cA = 0; cB = 0; }
  __syncthreads();
#pragma unroll
  for (int j = 0; j < 80; ++j) {
    if (__float_as_uint(ld[j]) < T) {
      int p = atomicAdd(&cA, 1);  // guaranteed < 255
      idxout[base + 1 + p] = t + j * 256;
    }
  }
  __syncthreads();
  int c1 = cA;
#pragma unroll
  for (int j = 0; j < 80; ++j) {
    if (__float_as_uint(ld[j]) == T) {
      int q = atomicAdd(&cB, 1);
      if (c1 + q < 255) idxout[base + 1 + c1 + q] = t + j * 256;
    }
  }
}

// ---------------- kNN stage 2: 32-NN of 512, iterative extraction ----------------

__global__ __launch_bounds__(256) void knn2_kernel(const float* __restrict__ xyz1, const float* __restrict__ kpts2,
                                                   int* __restrict__ idxout) {
  const int k = blockIdx.x, b = blockIdx.y, t = threadIdx.x;
  __shared__ float dist[512];
  __shared__ float rv[256];
  __shared__ int ri[256];
  const float* px = xyz1 + (size_t)b * 3 * NK1;
  const float* py = px + NK1;
  const float* pz = py + NK1;
  float kx = kpts2[(size_t)b * 3 * NK2 + k];
  float ky = kpts2[(size_t)b * 3 * NK2 + NK2 + k];
  float kz = kpts2[(size_t)b * 3 * NK2 + 2 * NK2 + k];
  for (int i = t; i < NK1; i += 256) {
    float dx = px[i] - kx, dy = py[i] - ky, dz = pz[i] - kz;
    dist[i] = dx * dx + dy * dy + dz * dz;
  }
  __syncthreads();
  const int base = ((b * NK2) + k) * NS2;
  for (int s = 0; s < NS2; ++s) {
    float bv = 3.0e38f;
    int bi = 0;
    for (int i = t; i < NK1; i += 256)
      if (dist[i] < bv) { bv = dist[i]; bi = i; }
    rv[t] = bv;
    ri[t] = bi;
    __syncthreads();
    for (int r = 128; r > 0; r >>= 1) {
      if (t < r && rv[t + r] < rv[t]) { rv[t] = rv[t + r]; ri[t] = ri[t + r]; }
      __syncthreads();
    }
    if (t == 0) {
      idxout[base + s] = ri[0];
      dist[ri[0]] = 3.0e38f;
    }
    __syncthreads();
  }
}

// ---------------- conv1: fused gather + features + MLP(8->64->64->256) + maxpool(256) ----------------
// one workgroup per (b, keypoint); wave w owns points [32w, 32w+32) through all layers (no inter-layer barrier)

__global__ __launch_bounds__(256) void conv1_kernel(const float* __restrict__ xyz, const float* __restrict__ feat,
                                                    const float* __restrict__ xyz1, const int* __restrict__ idx1,
                                                    const _Float16* __restrict__ wbuf, const float* __restrict__ scbuf,
                                                    const float* __restrict__ shbuf, float* __restrict__ f1out,
                                                    _Float16* __restrict__ f1h) {
  const int k = blockIdx.x & (NK1 - 1);
  const int b = blockIdx.x >> 9;
  const int t = threadIdx.x;
  const int lane = t & 31, wv = t >> 5;
  __shared__ alignas(16) _Float16 aIn[NS1][32];  // 16 KB  [point][channel]
  __shared__ alignas(16) _Float16 mid[NS1][64];  // 32 KB
  __shared__ unsigned poolbits[256];
  __shared__ float snc[3];

  // gather + feature construction (thread t == neighbor t)
  int pi = idx1[(((size_t)b * NK1) + k) * NS1 + t];
  const float* xb = xyz + (size_t)b * 3 * NPTS;
  float pxx = xb[pi], pyy = xb[NPTS + pi], pzz = xb[2 * NPTS + pi];
  float kx = xyz1[(size_t)b * 3 * NK1 + k];
  float ky = xyz1[(size_t)b * 3 * NK1 + NK1 + k];
  float kz = xyz1[(size_t)b * 3 * NK1 + 2 * NK1 + k];
  const float* fb = feat + (size_t)b * 4 * NPTS;
  float inten = fb[pi];
  float snx = fb[NPTS + pi], sny = fb[2 * NPTS + pi], snz = fb[3 * NPTS + pi];
  if (t == 0) { snc[0] = snx; snc[1] = sny; snc[2] = snz; }
  poolbits[t] = 0u;
  __syncthreads();
  float gx = pxx - kx, gy = pyy - ky, gz = pzz - kz;
  float dx = gx - kx, dy = gy - ky, dz = gz - kz;  // reference's d = gpc - xyz1 (double subtract)
  float scx = snc[0], scy = snc[1], scz = snc[2];
  float nr_d = angle3(scx, scy, scz, dx, dy, dz);
  float ni_d = angle3(snx, sny, snz, dx, dy, dz);
  float nr_ni = angle3(scx, scy, scz, snx, sny, snz);
  float dn2 = dx * dx + dy * dy + dz * dz;
  float dnorm = dn2 > 0.f ? sqrtf(dn2) : 0.f;
  aIn[t][0] = (_Float16)gx;
  aIn[t][1] = (_Float16)gy;
  aIn[t][2] = (_Float16)gz;
  aIn[t][3] = (_Float16)nr_d;
  aIn[t][4] = (_Float16)ni_d;
  aIn[t][5] = (_Float16)nr_ni;
  aIn[t][6] = (_Float16)dnorm;
  aIn[t][7] = (_Float16)inten;
#pragma unroll
  for (int c = 8; c < 32; ++c) aIn[t][c] = (_Float16)0.f;
  __syncthreads();

  const _Float16* w1g = wbuf + WO1;
  const _Float16* w2g = wbuf + WO2;
  const _Float16* w3g = wbuf + WO3;
  const float* sc1 = scbuf + SO1;  const float* sh1 = shbuf + SO1;
  const float* sc2 = scbuf + SO2;  const float* sh2 = shbuf + SO2;
  const float* sc3 = scbuf + SO3;  const float* sh3 = shbuf + SO3;

  // layer 1: K=32 (padded), M=64, N=32 per wave
  for (int nt = 0; nt < 2; ++nt) {
    int nbase = wv * 32 + nt * 16;
    v16h bm = load_b16(&aIn[0][0], nbase, 32, 0, lane);
#pragma unroll
    for (int mt = 0; mt < 4; ++mt) {
      v8f acc = {};
      acc = wmma_f16(load_a16(w1g, mt * 16, 32, 0, lane), bm, acc);
      store_bn_relu_f16(&mid[0][0], 64, nbase, mt * 16, acc, sc1, sh1, lane);
    }
  }
  // layer 2: 64 -> 64, in-place on mid (register-buffered)
  for (int nt = 0; nt < 2; ++nt) {
    int nbase = wv * 32 + nt * 16;
    v16h b0 = load_b16(&mid[0][0], nbase, 64, 0, lane);
    v16h b1 = load_b16(&mid[0][0], nbase, 64, 32, lane);
    v8f acc[4] = {};
#pragma unroll
    for (int mt = 0; mt < 4; ++mt) {
      acc[mt] = wmma_f16(load_a16(w2g, mt * 16, 64, 0, lane), b0, acc[mt]);
      acc[mt] = wmma_f16(load_a16(w2g, mt * 16, 64, 32, lane), b1, acc[mt]);
    }
#pragma unroll
    for (int mt = 0; mt < 4; ++mt)
      store_bn_relu_f16(&mid[0][0], 64, nbase, mt * 16, acc[mt], sc2, sh2, lane);
  }
  // layer 3: 64 -> 256, max-pool over neighbors via uint-bit atomic max (post-ReLU >= 0)
  for (int nt = 0; nt < 2; ++nt) {
    int nbase = wv * 32 + nt * 16;
    v16h b0 = load_b16(&mid[0][0], nbase, 64, 0, lane);
    v16h b1 = load_b16(&mid[0][0], nbase, 64, 32, lane);
    for (int mt = 0; mt < 16; ++mt) {
      v8f a = {};
      a = wmma_f16(load_a16(w3g, mt * 16, 64, 0, lane), b0, a);
      a = wmma_f16(load_a16(w3g, mt * 16, 64, 32, lane), b1, a);
      const int h = lane >> 4;
#pragma unroll
      for (int r = 0; r < 8; ++r) {
        int m = mt * 16 + r + 8 * h;
        float y = fmaxf(a[r] * sc3[m] + sh3[m], 0.f);
        atomicMax(&poolbits[m], __float_as_uint(y));
      }
    }
  }
  __syncthreads();
  float y = __uint_as_float(poolbits[t]);
  f1out[((size_t)b * 256 + t) * NK1 + k] = y;
  f1h[((size_t)b * 256 + t) * NK1 + k] = (_Float16)y;
}

// ---------------- conv2: fused gather + MLP(259->256->256->512) + maxpool(32) ----------------

__global__ __launch_bounds__(256) void conv2_kernel(const float* __restrict__ xyz1, const float* __restrict__ xyz2,
                                                    const int* __restrict__ idx2, const _Float16* __restrict__ f1h,
                                                    const _Float16* __restrict__ wbuf, const float* __restrict__ scbuf,
                                                    const float* __restrict__ shbuf, float* __restrict__ f2out,
                                                    _Float16* __restrict__ f2h) {
  const int k2 = blockIdx.x & (NK2 - 1);
  const int b = blockIdx.x >> 6;
  const int t = threadIdx.x;
  const int lane = t & 31, wv = t >> 5;
  __shared__ alignas(16) _Float16 aIn[NS2][288];   // 18 KB
  __shared__ alignas(16) _Float16 mid1[NS2][256];  // 16 KB
  __shared__ alignas(16) _Float16 mid2[NS2][256];  // 16 KB
  __shared__ unsigned poolbits[512];
  __shared__ int sidx[NS2];
  __shared__ float kcen[3];

  if (t < NS2) sidx[t] = idx2[(((size_t)b * NK2) + k2) * NS2 + t];
  if (t < 3) kcen[t] = xyz2[(size_t)b * 3 * NK2 + t * NK2 + k2];
  poolbits[t] = 0u;
  poolbits[t + 256] = 0u;
  __syncthreads();
  for (int f = t; f < NS2 * 288; f += 256) {
    int s = f / 288, c = f % 288;
    int pi = sidx[s];
    float v;
    if (c < 3)       v = xyz1[(size_t)b * 3 * NK1 + c * NK1 + pi] - kcen[c];
    else if (c < 259) v = (float)f1h[((size_t)b * 256 + (c - 3)) * NK1 + pi];
    else              v = 0.f;
    aIn[s][c] = (_Float16)v;
  }
  __syncthreads();

  const _Float16* w4g = wbuf + WO4;
  const _Float16* w5g = wbuf + WO5;
  const _Float16* w6g = wbuf + WO6;
  const float* sc4 = scbuf + SO4;  const float* sh4 = shbuf + SO4;
  const float* sc5 = scbuf + SO5;  const float* sh5 = shbuf + SO5;
  const float* sc6 = scbuf + SO6;  const float* sh6 = shbuf + SO6;

  // layer 4: K=288, wave w computes Mt {2w, 2w+1}
  for (int nt = 0; nt < 2; ++nt) {
    v8f acc[2] = {};
    for (int ks = 0; ks < 9; ++ks) {
      v16h bm = load_b16(&aIn[0][0], nt * 16, 288, ks * 32, lane);
#pragma unroll
      for (int u = 0; u < 2; ++u)
        acc[u] = wmma_f16(load_a16(w4g, (wv * 2 + u) * 16, 288, ks * 32, lane), bm, acc[u]);
    }
#pragma unroll
    for (int u = 0; u < 2; ++u)
      store_bn_relu_f16(&mid1[0][0], 256, nt * 16, (wv * 2 + u) * 16, acc[u], sc4, sh4, lane);
  }
  __syncthreads();
  // layer 5: K=256
  for (int nt = 0; nt < 2; ++nt) {
    v8f acc[2] = {};
    for (int ks = 0; ks < 8; ++ks) {
      v16h bm = load_b16(&mid1[0][0], nt * 16, 256, ks * 32, lane);
#pragma unroll
      for (int u = 0; u < 2; ++u)
        acc[u] = wmma_f16(load_a16(w5g, (wv * 2 + u) * 16, 256, ks * 32, lane), bm, acc[u]);
    }
#pragma unroll
    for (int u = 0; u < 2; ++u)
      store_bn_relu_f16(&mid2[0][0], 256, nt * 16, (wv * 2 + u) * 16, acc[u], sc5, sh5, lane);
  }
  __syncthreads();
  // layer 6: M=512, wave w computes Mt {4w..4w+3}; pool over 32 neighbors
  for (int nt = 0; nt < 2; ++nt) {
    v8f acc[4] = {};
    for (int ks = 0; ks < 8; ++ks) {
      v16h bm = load_b16(&mid2[0][0], nt * 16, 256, ks * 32, lane);
#pragma unroll
      for (int u = 0; u < 4; ++u)
        acc[u] = wmma_f16(load_a16(w6g, (wv * 4 + u) * 16, 256, ks * 32, lane), bm, acc[u]);
    }
    const int h = lane >> 4;
#pragma unroll
    for (int u = 0; u < 4; ++u) {
#pragma unroll
      for (int r = 0; r < 8; ++r) {
        int m = (wv * 4 + u) * 16 + r + 8 * h;
        float y = fmaxf(acc[u][r] * sc6[m] + sh6[m], 0.f);
        atomicMax(&poolbits[m], __float_as_uint(y));
      }
    }
  }
  __syncthreads();
  for (int c = t; c < 512; c += 256) {
    float y = __uint_as_float(poolbits[c]);
    f2out[((size_t)b * 512 + c) * NK2 + k2] = y;
    f2h[((size_t)b * NK2 + k2) * 512 + c] = (_Float16)y;
  }
}

// ---------------- conv3: MLP(512->512->512) + global maxpool(64) -> gf ----------------

__global__ __launch_bounds__(256) void conv3_kernel(const _Float16* __restrict__ f2h, const _Float16* __restrict__ wbuf,
                                                    const float* __restrict__ scbuf, const float* __restrict__ shbuf,
                                                    float* __restrict__ gf) {
  const int b = blockIdx.x, t = threadIdx.x;
  const int lane = t & 31, wv = t >> 5;
  __shared__ alignas(16) _Float16 mid[NK2][512];  // 64 KB
  const _Float16* w7g = wbuf + WO7;
  const _Float16* w8g = wbuf + WO8;
  const _Float16* bsrc = f2h + (size_t)b * NK2 * 512;  // [pt][ch]
  const float* sc7 = scbuf + SO7;  const float* sh7 = shbuf + SO7;
  const float* sc8 = scbuf + SO8;  const float* sh8 = shbuf + SO8;

  // layer 7: B from global f2h, wave w computes Mt {4w..4w+3}, 4 N tiles, K=512
  for (int nt = 0; nt < 4; ++nt) {
    v8f acc[4] = {};
    for (int ks = 0; ks < 16; ++ks) {
      v16h bm = load_b16(bsrc, nt * 16, 512, ks * 32, lane);
#pragma unroll
      for (int u = 0; u < 4; ++u)
        acc[u] = wmma_f16(load_a16(w7g, (wv * 4 + u) * 16, 512, ks * 32, lane), bm, acc[u]);
    }
#pragma unroll
    for (int u = 0; u < 4; ++u)
      store_bn_relu_f16(&mid[0][0], 512, nt * 16, (wv * 4 + u) * 16, acc[u], sc7, sh7, lane);
  }
  __syncthreads();
  // layer 8: B from LDS, pool over 64 keypoints into gf via global uint-bit atomic max (gf pre-zeroed)
  unsigned* gbits = (unsigned*)(gf + (size_t)b * 512);
  for (int nt = 0; nt < 4; ++nt) {
    v8f acc[4] = {};
    for (int ks = 0; ks < 16; ++ks) {
      v16h bm = load_b16(&mid[0][0], nt * 16, 512, ks * 32, lane);
#pragma unroll
      for (int u = 0; u < 4; ++u)
        acc[u] = wmma_f16(load_a16(w8g, (wv * 4 + u) * 16, 512, ks * 32, lane), bm, acc[u]);
    }
    const int h = lane >> 4;
#pragma unroll
    for (int u = 0; u < 4; ++u) {
#pragma unroll
      for (int r = 0; r < 8; ++r) {
        int m = (wv * 4 + u) * 16 + r + 8 * h;
        float y = fmaxf(acc[u][r] * sc8[m] + sh8[m], 0.f);
        atomicMax(&gbits[m], __float_as_uint(y));
      }
    }
  }
}

__global__ __launch_bounds__(256) void zero_kernel(float* __restrict__ p, int n) {
  int i = blockIdx.x * 256 + threadIdx.x;
  if (i < n) p[i] = 0.f;
}

// ---------------- host launcher ----------------

extern "C" void kernel_launch(void* const* d_in, const int* in_sizes, int n_in,
                              void* d_out, int out_size, void* d_ws, size_t ws_size,
                              hipStream_t stream) {
  (void)in_sizes; (void)n_in; (void)out_size; (void)ws_size;
  const float* xyz  = (const float*)d_in[0];
  const float* feat = (const float*)d_in[1];
  // params flattened per layer in pytree (sorted-key) order: W, b, beta, gamma, mean, var
  auto P = [&](int layer, int field) -> const float* { return (const float*)d_in[2 + layer * 6 + field]; };

  float* out = (float*)d_out;
  char* ws = (char*)d_ws;
  _Float16* wbuf = (_Float16*)(ws + WS_W);
  float* sc = (float*)(ws + WS_SC);
  float* sh = (float*)(ws + WS_SH);
  int* idx1 = (int*)(ws + WS_IDX1);
  int* idx2 = (int*)(ws + WS_IDX2);
  _Float16* f1h = (_Float16*)(ws + WS_F1H);
  _Float16* f2h = (_Float16*)(ws + WS_F2H);

  static const int LC[8][3] = {{64, 8, 32},    {64, 64, 64},   {256, 64, 64},  {256, 259, 288},
                               {256, 256, 256}, {512, 256, 256}, {512, 512, 512}, {512, 512, 512}};
  static const int WOFF[8] = {WO1, WO2, WO3, WO4, WO5, WO6, WO7, WO8};
  static const int SOFF[8] = {SO1, SO2, SO3, SO4, SO5, SO6, SO7, SO8};
  for (int l = 0; l < 8; ++l) {
    int cout = LC[l][0], cin = LC[l][1], cp = LC[l][2];
    int tot = cout * cp;
    prep_layer<<<(tot + 255) / 256, 256, 0, stream>>>(P(l, 0), P(l, 1), P(l, 2), P(l, 3), P(l, 4), P(l, 5),
                                                      wbuf + WOFF[l], sc + SOFF[l], sh + SOFF[l], cout, cin, cp);
  }

  fps_kernel<1024, 20><<<NB, 1024, 0, stream>>>(xyz, NPTS, NK1, out + OUT_XYZ1);
  fps_kernel<512, 1><<<NB, 512, 0, stream>>>(out + OUT_XYZ1, NK1, NK2, out + OUT_XYZ2);

  knn1_kernel<<<dim3(NK1, NB), 256, 0, stream>>>(xyz, out + OUT_XYZ1, idx1);
  knn2_kernel<<<dim3(NK2, NB), 256, 0, stream>>>(out + OUT_XYZ1, out + OUT_XYZ2, idx2);

  conv1_kernel<<<NB * NK1, 256, 0, stream>>>(xyz, feat, out + OUT_XYZ1, idx1, wbuf, sc, sh, out + OUT_F1, f1h);
  conv2_kernel<<<NB * NK2, 256, 0, stream>>>(out + OUT_XYZ1, out + OUT_XYZ2, idx2, f1h, wbuf, sc, sh,
                                             out + OUT_F2, f2h);
  zero_kernel<<<16, 256, 0, stream>>>(out + OUT_GF, NB * 512);
  conv3_kernel<<<NB, 256, 0, stream>>>(f2h, wbuf, sc, sh, out + OUT_GF);
}
